// Attention_7464653160993
// MI455X (gfx1250) — compile-verified, matrix operand order checked by reading, not run on hardware
//
#include <hip/hip_runtime.h>
#include <hip/hip_bf16.h>

// Problem constants
#define BB 4
#define TT 2048
#define EE 2048
#define FF 6144   // 3*EE
#define HH 16
#define DD 128

typedef __attribute__((ext_vector_type(16))) __bf16 v16bf;
typedef __attribute__((ext_vector_type(8)))  __bf16 v8bf;
typedef __attribute__((ext_vector_type(8)))  float  v8f;
typedef __attribute__((ext_vector_type(4)))  unsigned int v4u;
typedef __attribute__((ext_vector_type(8)))  int v8i;
typedef __attribute__((ext_vector_type(4)))  int v4i;

// ---------------------------------------------------------------------------
// WMMA helper
// ---------------------------------------------------------------------------
static __device__ __forceinline__ v8f wmma_bf16(v16bf a, v16bf b, v8f c) {
  return __builtin_amdgcn_wmma_f32_16x16x32_bf16(
      /*neg_a=*/false, a, /*neg_b=*/false, b,
      /*c_mod=*/(short)0, c, /*reuse_a=*/false, /*reuse_b=*/false);
}

// Load a 16(row) x 32(K) bf16 fragment, rows contiguous with leading dim `ld`.
// ISA 7.12.2 A-layout: lane m (0..15), half c=lane>>4 holds K=c*8..+7 (elems
// 0..7) and K=16+c*8..+7 (elems 8..15) of row m.  Works for global or LDS ptrs.
static __device__ __forceinline__ v16bf load_frag_rowK(const __bf16* __restrict__ base,
                                                       int ld, int lane) {
  const int m = lane & 15;
  const int c = lane >> 4;
  const __bf16* r = base + m * ld + c * 8;
  v8bf lo = *(const v8bf*)(r);
  v8bf hi = *(const v8bf*)(r + 16);
  v16bf f;
#pragma unroll
  for (int i = 0; i < 8; ++i) { f[i] = lo[i]; f[i + 8] = hi[i]; }
  return f;
}

// ---------------------------------------------------------------------------
// Tensor Data Mover: issue a 2-D tile load (bf16 elements) into LDS.
// Builds D# group0/group1 per CDNA5 ISA §8.3/§8.4; groups 2/3 unused (2-D).
// 6-arg builtin (clang-23 / therock-10.0 form): g0, g1, g2, g3, extra, cpol.
//   tile_d0  : contiguous (innermost) tile extent, elements
//   tile_d1  : number of rows
//   stride0  : row-to-row stride in elements
// ---------------------------------------------------------------------------
#if __has_builtin(__builtin_amdgcn_tensor_load_to_lds)
#define HAVE_TDM 1
static __device__ __forceinline__ void tdm_load_2d(const __bf16* gptr,
                                                   unsigned lds_off,
                                                   unsigned tile_d0, unsigned tile_d1,
                                                   unsigned tensor_d0, unsigned tensor_d1,
                                                   unsigned long long stride0) {
  const unsigned long long ga = (unsigned long long)(uintptr_t)gptr;
  v4u g0;
  g0[0] = 1u;                                              // count=1, user D#
  g0[1] = lds_off;                                         // lds_addr (bytes)
  g0[2] = (unsigned)(ga & 0xFFFFFFFFu);                    // global_addr[31:0]
  g0[3] = (unsigned)((ga >> 32) & 0x01FFFFFFu) | (2u << 30); // addr[56:32] | type=2
  v8i g1;
  g1[0] = (int)(1u << 16);                                 // data_size=1 (2B), mask=0
  g1[1] = (int)((tensor_d0 & 0xFFFFu) << 16);              // tensor_dim0 lo16 (abar=0)
  g1[2] = (int)((tensor_d0 >> 16) | ((tensor_d1 & 0xFFFFu) << 16)); // dim0 hi | dim1 lo
  g1[3] = (int)((tensor_d1 >> 16) | (tile_d0 << 16));      // dim1 hi | tile_dim0
  g1[4] = (int)(tile_d1 & 0xFFFFu);                        // tile_dim1 | tile_dim2=0
  g1[5] = (int)(unsigned)(stride0 & 0xFFFFFFFFull);        // dim0_stride lo32
  g1[6] = (int)(unsigned)((stride0 >> 32) & 0xFFFFull);    // dim0_stride hi16 | dim1_stride lo16=0
  g1[7] = 0;                                               // dim1_stride hi32 (unused, 2-D)
  const v4i zero4 = (v4i)0;
  const v8i zero8 = (v8i)0;
  __builtin_amdgcn_tensor_load_to_lds(g0, g1, zero4, zero4, zero8, 0);
}
#else
#define HAVE_TDM 0
#endif

// ---------------------------------------------------------------------------
// f32 -> bf16 conversion (grid-stride, vectorized x4; all sizes %4 == 0)
// ---------------------------------------------------------------------------
__global__ __launch_bounds__(256) void cvt_bf16_kernel(const float* __restrict__ in,
                                                       __bf16* __restrict__ out, int n) {
  int stride4 = (int)(gridDim.x * blockDim.x) * 4;
  for (int j = (int)(blockIdx.x * blockDim.x + threadIdx.x) * 4; j < n; j += stride4) {
    float4 v = *(const float4*)(in + j);
    out[j + 0] = (__bf16)v.x;
    out[j + 1] = (__bf16)v.y;
    out[j + 2] = (__bf16)v.z;
    out[j + 3] = (__bf16)v.w;
  }
}

// ---------------------------------------------------------------------------
// GEMM1: qkv[m, f] = sum_e xb[m, e] * w1b[f, e]
//   M=8192, N=6144, K=2048; 8 waves (4x2), wave tile 32x64.
//   Each 128-wide N block lies inside exactly ONE (s,h) slab -> uniform epilogue.
// ---------------------------------------------------------------------------
__global__ __launch_bounds__(256) void gemm_qkv_kernel(const __bf16* __restrict__ xb,
                                                       const __bf16* __restrict__ w1b,
                                                       __bf16* __restrict__ qarr,
                                                       __bf16* __restrict__ karr,
                                                       __bf16* __restrict__ vtarr) {
  const int lane = threadIdx.x & 31;
  const int wid  = threadIdx.x >> 5;
  const int wm   = wid >> 1;              // 0..3
  const int wn   = wid & 1;               // 0..1
  const int m0   = blockIdx.y * 128 + wm * 32;
  const int n0   = blockIdx.x * 128 + wn * 64;

  v8f acc[2][4];
#pragma unroll
  for (int i = 0; i < 2; ++i)
#pragma unroll
    for (int j = 0; j < 4; ++j) acc[i][j] = (v8f)(0.0f);

  for (int k0 = 0; k0 < EE; k0 += 32) {
    __builtin_prefetch(xb  + (size_t)m0 * EE + k0 + 64, 0, 1);
    __builtin_prefetch(w1b + (size_t)n0 * EE + k0 + 64, 0, 1);
    v16bf a[2], b[4];
#pragma unroll
    for (int i = 0; i < 2; ++i)
      a[i] = load_frag_rowK(xb + (size_t)(m0 + 16 * i) * EE + k0, EE, lane);
#pragma unroll
    for (int j = 0; j < 4; ++j)
      b[j] = load_frag_rowK(w1b + (size_t)(n0 + 16 * j) * EE + k0, EE, lane);
#pragma unroll
    for (int i = 0; i < 2; ++i)
#pragma unroll
      for (int j = 0; j < 4; ++j)
        acc[i][j] = wmma_bf16(a[i], b[j], acc[i][j]);
  }

  // Uniform (block-wide) q/k/v selector and head.
  const int nblk = blockIdx.x * 128;
  const int s    = nblk >> 11;            // 0=q 1=k 2=v
  const int h    = (nblk >> 7) & 15;
  const int dof  = wn * 64;               // d offset of this wave inside the head
  const int nn   = lane & 15;
  const int half = lane >> 4;

  if (s < 2) {
    __bf16* __restrict__ dst = (s == 0) ? qarr : karr;
#pragma unroll
    for (int i = 0; i < 2; ++i)
#pragma unroll
      for (int r = 0; r < 8; ++r) {
        const int m  = m0 + i * 16 + half * 8 + r;
        const int bh = (m >> 11) * HH + h;
        const size_t row = (size_t)bh * TT + (m & 2047);
#pragma unroll
        for (int j = 0; j < 4; ++j)
          dst[row * DD + dof + j * 16 + nn] = (__bf16)acc[i][j][r];
      }
  } else {
#pragma unroll
    for (int i = 0; i < 2; ++i)
#pragma unroll
      for (int r = 0; r < 8; ++r) {
        const int m  = m0 + i * 16 + half * 8 + r;
        const int bh = (m >> 11) * HH + h;
        const int t  = m & 2047;
#pragma unroll
        for (int j = 0; j < 4; ++j) {
          const int d = dof + j * 16 + nn;
          vtarr[((size_t)bh * DD + d) * TT + t] = (__bf16)acc[i][j][r];
        }
      }
  }
}

// ---------------------------------------------------------------------------
// Flash attention: grid = (T/128, B*H); 8 waves, wave owns 16 q rows.
// Wave 0 drives the Tensor Data Mover to double-buffer the shared 32-key
// K tile (32x128) and Vt tile (128x32) in LDS; all waves consume via ds loads.
// ---------------------------------------------------------------------------
__global__ __launch_bounds__(256) void attn_kernel(const __bf16* __restrict__ qarr,
                                                   const __bf16* __restrict__ karr,
                                                   const __bf16* __restrict__ vtarr,
                                                   __bf16* __restrict__ ob) {
  __shared__ __bf16 kbuf[2][32 * 128];   // 2 x 8 KB
  __shared__ __bf16 vbuf[2][128 * 32];   // 2 x 8 KB
  __shared__ __bf16 plds[8 * 16 * 32];   // per-wave P staging, 8 KB

  const int lane  = threadIdx.x & 31;
  const int wid   = threadIdx.x >> 5;
  const int bh    = blockIdx.y;          // b*16 + h
  const int qrow0 = blockIdx.x * 128 + wid * 16;

  const __bf16* Q  = qarr  + (size_t)bh * TT * DD;
  const __bf16* Kp = karr  + (size_t)bh * TT * DD;
  const __bf16* Vt = vtarr + (size_t)bh * DD * TT;
  __bf16* pl = plds + wid * (16 * 32);

  const float scale = 0.08838834764831845f;   // 1/sqrt(128)

  // Resident Q fragments: 16 rows x 128 d (4 K-chunks of 32)
  v16bf qf[4];
#pragma unroll
  for (int c4 = 0; c4 < 4; ++c4)
    qf[c4] = load_frag_rowK(Q + (size_t)qrow0 * DD + c4 * 32, DD, lane);

  v8f o[8];
#pragma unroll
  for (int f = 0; f < 8; ++f) o[f] = (v8f)(0.0f);
  float mrow[8], lrow[8];
#pragma unroll
  for (int r = 0; r < 8; ++r) { mrow[r] = -3.0e38f; lrow[r] = 0.0f; }

  const int nn   = lane & 15;
  const int half = lane >> 4;

#if HAVE_TDM
  const unsigned kofs[2] = {(unsigned)(uintptr_t)&kbuf[0][0],
                            (unsigned)(uintptr_t)&kbuf[1][0]};
  const unsigned vofs[2] = {(unsigned)(uintptr_t)&vbuf[0][0],
                            (unsigned)(uintptr_t)&vbuf[1][0]};
  if (wid == 0) {   // prologue: stage block 0 into buffer 0
    tdm_load_2d(Kp, kofs[0], /*tile*/ DD, 32, /*tensor*/ DD, TT, /*stride*/ DD);
    tdm_load_2d(Vt, vofs[0], /*tile*/ 32, DD, /*tensor*/ TT, DD, /*stride*/ TT);
  }
#endif

  const int nblocks = TT / 32;
  for (int jb = 0; jb < nblocks; ++jb) {
    const int j0  = jb * 32;
    const int buf = jb & 1;

#if HAVE_TDM
    if (wid == 0) {
      if (jb + 1 < nblocks) {
        const int j1 = j0 + 32;
        tdm_load_2d(Kp + (size_t)j1 * DD, kofs[buf ^ 1], DD, 32, DD, TT, DD);
        tdm_load_2d(Vt + j1,              vofs[buf ^ 1], 32, DD, TT, DD, TT);
        __builtin_amdgcn_s_wait_tensorcnt((short)2);   // current buffer complete
      } else {
        __builtin_amdgcn_s_wait_tensorcnt((short)0);
      }
    }
    __syncthreads();
    const __bf16* kb = kbuf[buf];
    const __bf16* vb = vbuf[buf];
#else
    const __bf16* kb = Kp + (size_t)j0 * DD;   // fallback: straight from global
    const __bf16* vb = nullptr;
#endif

    // ---- S = Q K^T for 32 keys ----
    v8f s0 = (v8f)(0.0f), s1 = (v8f)(0.0f);
#pragma unroll
    for (int c4 = 0; c4 < 4; ++c4) {
      v16bf kf0 = load_frag_rowK(kb + c4 * 32, DD, lane);
      v16bf kf1 = load_frag_rowK(kb + 16 * DD + c4 * 32, DD, lane);
      s0 = wmma_bf16(qf[c4], kf0, s0);
      s1 = wmma_bf16(qf[c4], kf1, s1);
    }

    // ---- online softmax ----
    float alpha[8];
#pragma unroll
    for (int r = 0; r < 8; ++r) {
      float v0 = s0[r] * scale;
      float v1 = s1[r] * scale;
      float mx = fmaxf(v0, v1);
#pragma unroll
      for (int off = 1; off < 16; off <<= 1)
        mx = fmaxf(mx, __shfl_xor(mx, off, 32));
      float mnew = fmaxf(mrow[r], mx);
      alpha[r] = __expf(mrow[r] - mnew);
      mrow[r] = mnew;
      float p0 = __expf(v0 - mnew);
      float p1 = __expf(v1 - mnew);
      s0[r] = p0; s1[r] = p1;
      float rs = p0 + p1;
#pragma unroll
      for (int off = 1; off < 16; off <<= 1)
        rs += __shfl_xor(rs, off, 32);
      lrow[r] = lrow[r] * alpha[r] + rs;
    }
#pragma unroll
    for (int f = 0; f < 8; ++f)
#pragma unroll
      for (int r = 0; r < 8; ++r) o[f][r] *= alpha[r];

    // ---- transpose P (C-layout) -> A-layout via LDS ----
#pragma unroll
    for (int r = 0; r < 8; ++r) {
      const int row = half * 8 + r;
      pl[row * 32 + nn]      = (__bf16)s0[r];
      pl[row * 32 + 16 + nn] = (__bf16)s1[r];
    }
    __syncthreads();
    v16bf pf = load_frag_rowK(pl, 32, lane);

    // ---- O += P @ V ----
#pragma unroll
    for (int f = 0; f < 8; ++f) {
#if HAVE_TDM
      v16bf vf = load_frag_rowK(vb + (f * 16) * 32, 32, lane);
#else
      v16bf vf = load_frag_rowK(Vt + (size_t)(f * 16) * TT + j0, TT, lane);
#endif
      o[f] = wmma_bf16(pf, vf, o[f]);
    }
    __syncthreads();   // all reads of this buffer done before next TDM issue
  }

  // ---- normalize and store O as bf16 [B*T, E], col = h*128 + d ----
  float inv[8];
#pragma unroll
  for (int r = 0; r < 8; ++r) inv[r] = 1.0f / lrow[r];
  const int b = bh >> 4;
  const int h = bh & 15;
#pragma unroll
  for (int f = 0; f < 8; ++f) {
    const int col = h * 128 + f * 16 + nn;
#pragma unroll
    for (int r = 0; r < 8; ++r) {
      const int row = b * TT + qrow0 + half * 8 + r;
      ob[(size_t)row * EE + col] = (__bf16)(o[f][r] * inv[r]);
    }
  }
}

// ---------------------------------------------------------------------------
// GEMM2: out[m, n] = sum_e ob[m, e] * w2b[n, e]   (M=8192, N=2048, K=2048)
// ---------------------------------------------------------------------------
__global__ __launch_bounds__(256) void gemm_out_kernel(const __bf16* __restrict__ ob,
                                                       const __bf16* __restrict__ w2b,
                                                       float* __restrict__ out) {
  const int lane = threadIdx.x & 31;
  const int wid  = threadIdx.x >> 5;
  const int wm   = wid >> 1;
  const int wn   = wid & 1;
  const int m0   = blockIdx.y * 128 + wm * 32;
  const int n0   = blockIdx.x * 128 + wn * 64;

  v8f acc[2][4];
#pragma unroll
  for (int i = 0; i < 2; ++i)
#pragma unroll
    for (int j = 0; j < 4; ++j) acc[i][j] = (v8f)(0.0f);

  for (int k0 = 0; k0 < EE; k0 += 32) {
    __builtin_prefetch(ob  + (size_t)m0 * EE + k0 + 64, 0, 1);
    __builtin_prefetch(w2b + (size_t)n0 * EE + k0 + 64, 0, 1);
    v16bf a[2], b[4];
#pragma unroll
    for (int i = 0; i < 2; ++i)
      a[i] = load_frag_rowK(ob + (size_t)(m0 + 16 * i) * EE + k0, EE, lane);
#pragma unroll
    for (int j = 0; j < 4; ++j)
      b[j] = load_frag_rowK(w2b + (size_t)(n0 + 16 * j) * EE + k0, EE, lane);
#pragma unroll
    for (int i = 0; i < 2; ++i)
#pragma unroll
      for (int j = 0; j < 4; ++j)
        acc[i][j] = wmma_bf16(a[i], b[j], acc[i][j]);
  }

  const int nn   = lane & 15;
  const int half = lane >> 4;
#pragma unroll
  for (int i = 0; i < 2; ++i)
#pragma unroll
    for (int r = 0; r < 8; ++r) {
      const int m = m0 + i * 16 + half * 8 + r;
#pragma unroll
      for (int j = 0; j < 4; ++j)
        out[(size_t)m * EE + n0 + j * 16 + nn] = acc[i][j][r];
    }
}

// ---------------------------------------------------------------------------
// Launch
// ---------------------------------------------------------------------------
extern "C" void kernel_launch(void* const* d_in, const int* in_sizes, int n_in,
                              void* d_out, int out_size, void* d_ws, size_t ws_size,
                              hipStream_t stream) {
  const float* x  = (const float*)d_in[0];   // [4, 2048, 2048]
  const float* w1 = (const float*)d_in[1];   // [6144, 2048]
  const float* w2 = (const float*)d_in[2];   // [2048, 2048]
  float* out = (float*)d_out;                // [4, 2048, 2048]

  char* ws = (char*)d_ws;
  const size_t XB_BYTES  = (size_t)BB * TT * EE * 2;   // 32 MB
  const size_t W1B_BYTES = (size_t)FF * EE * 2;        // 24 MB
  const size_t W2B_BYTES = (size_t)EE * EE * 2;        //  8 MB
  const size_t QKV_BYTES = (size_t)BB * TT * EE * 2;   // 32 MB each

  __bf16* xb    = (__bf16*)(ws);
  __bf16* w1b   = (__bf16*)(ws + XB_BYTES);
  __bf16* w2b   = (__bf16*)(ws + XB_BYTES + W1B_BYTES);
  __bf16* qarr  = (__bf16*)(ws + XB_BYTES + W1B_BYTES + W2B_BYTES);
  __bf16* karr  = (__bf16*)(ws + XB_BYTES + W1B_BYTES + W2B_BYTES + QKV_BYTES);
  __bf16* vtarr = (__bf16*)(ws + XB_BYTES + W1B_BYTES + W2B_BYTES + 2 * QKV_BYTES);
  __bf16* ob    = (__bf16*)(ws + XB_BYTES + W1B_BYTES + W2B_BYTES + 3 * QKV_BYTES);

  cvt_bf16_kernel<<<4096, 256, 0, stream>>>(x,  xb,  BB * TT * EE);
  cvt_bf16_kernel<<<3072, 256, 0, stream>>>(w1, w1b, FF * EE);
  cvt_bf16_kernel<<<1024, 256, 0, stream>>>(w2, w2b, EE * EE);

  gemm_qkv_kernel<<<dim3(FF / 128, (BB * TT) / 128), 256, 0, stream>>>(
      xb, w1b, qarr, karr, vtarr);

  attn_kernel<<<dim3(TT / 128, BB * HH), 256, 0, stream>>>(qarr, karr, vtarr, ob);

  gemm_out_kernel<<<dim3(EE / 128, (BB * TT) / 128), 256, 0, stream>>>(ob, w2b, out);
}
